// MGR_17609365914202
// MI455X (gfx1250) — matrix operand order repaired and checked by reading, not compiled, not run on hardware
//
#include <hip/hip_runtime.h>
#include <hip/hip_bf16.h>
#include <stdint.h>

// Problem constants (match reference)
#define BB   8
#define NN   8192
#define CCH  256
#define HH   8
#define KCH  32
#define VCH  32
#define CHT  320          // stacked proj channels: 256 q | 32 k | 32 v
#define EPSB 1e-5f

typedef __bf16 bf16;
typedef bf16    v16bf  __attribute__((ext_vector_type(16)));
typedef float   v8f    __attribute__((ext_vector_type(8)));
typedef uint32_t u32x4 __attribute__((ext_vector_type(4)));

union AccU  { v8f v;  float f[8]; };
union FragU { v16bf v; u32x4 q[2]; };

static __device__ inline v8f wmma_bf16(v16bf a, v16bf b, v8f c) {
  // D = A(16x32 bf16) * B(32x16 bf16) + C(16x16 f32)
  return __builtin_amdgcn_wmma_f32_16x16x32_bf16(
      false, a, false, b, (short)0, c, false, false);
}

// Async global->LDS copy, 16 bytes per lane, tracked by ASYNCcnt.
// LDS byte address = low 32 bits of the generic __shared__ pointer
// (generic LDS pointers carry the wave-relative offset in addr[31:0]).
static __device__ __forceinline__ void asy_b128(void* lds, const void* g) {
  asm volatile("global_load_async_to_lds_b128 %0, %1, off"
               :: "v"((uint32_t)(uintptr_t)lds), "v"(g) : "memory");
}
static __device__ __forceinline__ void asy_wait0() {
  asm volatile("s_wait_asynccnt 0" ::: "memory");
}

// ---------------------------------------------------------------------------
// K0: cast/pack weights to bf16.
//   Wall[320][256] = {Wq; Wk; Wv},  Wsb[256][256],  A2t[(k*32+w)][v] = A[w][v][k]
// ---------------------------------------------------------------------------
__global__ void k_prep(const float* __restrict__ Wq, const float* __restrict__ Wk,
                       const float* __restrict__ Wv, const float* __restrict__ Ws,
                       const float* __restrict__ A,
                       bf16* __restrict__ Wall, bf16* __restrict__ Wsb,
                       bf16* __restrict__ A2t)
{
  int i = blockIdx.x * 256 + threadIdx.x;
  if (i < CHT * 256) {
    int r = i >> 8, c = i & 255;
    float f = (r < 256) ? Wq[i]
            : (r < 288) ? Wk[(r - 256) * 256 + c]
                        : Wv[(r - 288) * 256 + c];
    Wall[i] = (bf16)f;
  }
  int j = i - CHT * 256;
  if (j >= 0 && j < 256 * 256) Wsb[j] = (bf16)Ws[j];
  int a = i - CHT * 256 - 256 * 256;
  if (a >= 0 && a < 1024 * 32) {
    int col = a >> 5, v = a & 31;
    int kc = col >> 5, w = col & 31;            // col = k*32 + w
    A2t[a] = (bf16)A[w * 1024 + v * 32 + kc];   // A[w][v][k]
  }
}

// ---------------------------------------------------------------------------
// K0b: transpose x to bf16 "GEMM-B layout":  xt[b][n][c]
// 32x32 LDS tile transpose; one 96 MB pass, L2-resident afterwards.
// ---------------------------------------------------------------------------
__global__ __launch_bounds__(256) void k_xt(const float* __restrict__ x,
                                            bf16* __restrict__ xt)
{
  __shared__ float t[32][33];
  const int b  = blockIdx.z;
  const int n0 = blockIdx.x * 32, c0 = blockIdx.y * 32;
  const int tx = threadIdx.x & 31, ty = threadIdx.x >> 5;   // 8 row-groups
  for (int r = ty; r < 32; r += 8)
    t[r][tx] = x[((size_t)b * CCH + c0 + r) * NN + n0 + tx];
  __syncthreads();
  for (int r = ty; r < 32; r += 8)
    xt[((size_t)b * NN + n0 + r) * CCH + c0 + tx] = (bf16)t[tx][r];
}

// ---------------------------------------------------------------------------
// K1: unified WMMA GEMM   W[Mtot,256] x Bsrc[b][n][256]^T -> out[b][Mtot][N]
// block 256 thr = 8 waves; tile 64 rows x 128 cols; wave = 16x64 (4 C tiles).
// B chunks staged via async-to-LDS with ping-pong double buffering.
// Used for the projection GEMM (Mtot=320, Bsrc=xt) and skip GEMM (Mtot=256,
// Bsrc=Y) -- Y is produced directly in [n][c] layout so no transpose needed.
// ---------------------------------------------------------------------------
__global__ __launch_bounds__(256) void k_gemm(
    const bf16* __restrict__ Bsrc, const bf16* __restrict__ W,
    float* __restrict__ out, int Mtot)
{
  __shared__ bf16 sW[64 * 256];        // 32 KB, [m][k] row-major
  __shared__ bf16 sX[2][128 * 32];     // 2 x 8 KB, [n][k] ping-pong
  const int tid  = threadIdx.x;
  const int wave = tid >> 5, lane = tid & 31;
  const int m0 = blockIdx.x * 64;
  const int n0 = blockIdx.y * 128;
  const int b  = blockIdx.z;
  const bf16* bbase = Bsrc + ((size_t)b * NN + n0) * 256;

  // async-stage the whole weight tile (64x256) + B chunk 0
  for (int i = tid; i < 2048; i += 256)
    asy_b128(sW + i * 8, W + (size_t)m0 * 256 + i * 8);
  for (int i = tid; i < 512; i += 256) {
    int nn = i >> 2, j = (i & 3) * 8;
    asy_b128(&sX[0][nn * 32 + j], bbase + (size_t)nn * 256 + j);
  }
  asy_wait0();
  __syncthreads();

  const int mrow = (wave & 3) * 16;
  const int ncol = (wave >> 2) * 64;
  AccU acc[4];
#pragma unroll
  for (int t = 0; t < 4; ++t)
#pragma unroll
    for (int r = 0; r < 8; ++r) acc[t].f[r] = 0.f;

  for (int kc = 0; kc < 8; ++kc) {
    const int cur = kc & 1;
    if (kc < 7) {  // issue next chunk while computing this one
      const bf16* src = bbase + (kc + 1) * 32;
      for (int i = tid; i < 512; i += 256) {
        int nn = i >> 2, j = (i & 3) * 8;
        asy_b128(&sX[1 - cur][nn * 32 + j], src + (size_t)nn * 256 + j);
      }
    }
    FragU fa;
    { // A fragment: M=lane&15; lane<16 -> K 0..7,16..23 ; else 8..15,24..31
      int row = mrow + (lane & 15);
      int hb  = row * 256 + kc * 32 + ((lane < 16) ? 0 : 8);
      fa.q[0] = *(const u32x4*)(sW + hb);
      fa.q[1] = *(const u32x4*)(sW + hb + 16);
    }
#pragma unroll
    for (int t = 0; t < 4; ++t) {
      FragU fb;  // B fragment: N=lane&15; lane<16 -> K 0..7,16..23 ; else 8..15,24..31
      int col = ncol + t * 16 + (lane & 15);
      int hb  = col * 32 + ((lane < 16) ? 0 : 8);
      fb.q[0] = *(const u32x4*)(&sX[cur][hb]);
      fb.q[1] = *(const u32x4*)(&sX[cur][hb + 16]);
      acc[t].v = wmma_bf16(fa.v, fb.v, acc[t].v);
    }
    asy_wait0();
    __syncthreads();
  }
  // store f32; C layout: M = r + 8*(lane>>4), N = lane&15
#pragma unroll
  for (int t = 0; t < 4; ++t) {
    int nidx  = n0 + ncol + t * 16 + (lane & 15);
    int mbase = m0 + mrow + ((lane >> 4) << 3);
#pragma unroll
    for (int r = 0; r < 8; ++r)
      out[((size_t)b * Mtot + mbase + r) * NN + nidx] = acc[t].f[r];
  }
}

// ---------------------------------------------------------------------------
// K2: BatchNorm stats over (B, N) for one channel per block.
// writes fused scale/shift: y = raw*sc[c] + sh[c]
// ---------------------------------------------------------------------------
__global__ __launch_bounds__(256) void k_bnstats(
    const float* __restrict__ data, int chTotal, int chOff,
    const float* __restrict__ gamma, const float* __restrict__ beta,
    float* __restrict__ sc, float* __restrict__ sh)
{
  __shared__ float rs[256], rq[256];
  const int c = blockIdx.x, tid = threadIdx.x;
  float s = 0.f, q = 0.f;
  for (int b = 0; b < BB; ++b) {
    const float* p = data + ((size_t)b * chTotal + chOff + c) * NN;
    for (int n = tid; n < NN; n += 256) { float v = p[n]; s += v; q += v * v; }
  }
  rs[tid] = s; rq[tid] = q; __syncthreads();
  for (int st = 128; st > 0; st >>= 1) {
    if (tid < st) { rs[tid] += rs[tid + st]; rq[tid] += rq[tid + st]; }
    __syncthreads();
  }
  if (tid == 0) {
    float inv = 1.f / (float)(BB * NN);
    float m   = rs[0] * inv;
    float var = rq[0] * inv - m * m;
    float scale = gamma[c] * rsqrtf(var + EPSB);
    sc[c] = scale;
    sh[c] = beta[c] - m * scale;
  }
}

// ---------------------------------------------------------------------------
// K3: softmax stats per (b, k-channel) row (max, 1/sum(exp))
// ---------------------------------------------------------------------------
__global__ __launch_bounds__(256) void k_softmax_stats(
    const float* __restrict__ qkv, float* __restrict__ kmax,
    float* __restrict__ kinv)
{
  __shared__ float red[256];
  const int bk = blockIdx.x, tid = threadIdx.x;
  const int b = bk >> 5, kc = bk & 31;
  const float* p = qkv + ((size_t)b * CHT + 256 + kc) * NN;
  float mx = -3.0e38f;
  for (int n = tid; n < NN; n += 256) mx = fmaxf(mx, p[n]);
  red[tid] = mx; __syncthreads();
  for (int st = 128; st > 0; st >>= 1) {
    if (tid < st) red[tid] = fmaxf(red[tid], red[tid + st]);
    __syncthreads();
  }
  mx = red[0]; __syncthreads();
  float s = 0.f;
  for (int n = tid; n < NN; n += 256) s += __expf(p[n] - mx);
  red[tid] = s; __syncthreads();
  for (int st = 128; st > 0; st >>= 1) {
    if (tid < st) red[tid] += red[tid + st];
    __syncthreads();
  }
  if (tid == 0) { kmax[bk] = mx; kinv[bk] = 1.f / red[0]; }
}

// ---------------------------------------------------------------------------
// K4: kv[b][k][v] = sum_n softmax(k)[b,k,n] * vbn[b,v,n]   (block per (b,k))
// ---------------------------------------------------------------------------
__global__ __launch_bounds__(256) void k_kv(
    const float* __restrict__ qkv, const float* __restrict__ kmax,
    const float* __restrict__ kinv, const float* __restrict__ vs,
    const float* __restrict__ vh, float* __restrict__ kvout)
{
  __shared__ float red[256];
  const int bk = blockIdx.x, tid = threadIdx.x;
  const int b = bk >> 5, kc = bk & 31;
  const float* kp = qkv + ((size_t)b * CHT + 256 + kc) * NN;
  const float* vb = qkv + ((size_t)b * CHT + 288) * NN;
  const float mx = kmax[bk], inv = kinv[bk];
  float acc[32];
#pragma unroll
  for (int v = 0; v < 32; ++v) acc[v] = 0.f;
  for (int n = tid; n < NN; n += 256) {
    float kn = __expf(kp[n] - mx) * inv;
#pragma unroll
    for (int v = 0; v < 32; ++v)
      acc[v] += kn * (vb[(size_t)v * NN + n] * vs[v] + vh[v]);
  }
  for (int v = 0; v < 32; ++v) {
    red[tid] = acc[v]; __syncthreads();
    for (int st = 128; st > 0; st >>= 1) {
      if (tid < st) red[tid] += red[tid + st];
      __syncthreads();
    }
    if (tid == 0) kvout[(size_t)bk * 32 + v] = red[0];
    __syncthreads();
  }
}

// ---------------------------------------------------------------------------
// K5: vbn transposed to bf16:  vtb[b][n][v]
// ---------------------------------------------------------------------------
__global__ void k_vtb(const float* __restrict__ qkv, const float* __restrict__ vs,
                      const float* __restrict__ vh, bf16* __restrict__ vtb)
{
  size_t i = (size_t)blockIdx.x * 256 + threadIdx.x;
  if (i >= (size_t)BB * NN * 32) return;
  int v = (int)(i & 31);
  size_t nb = i >> 5;
  int n = (int)(nb % NN);
  int b = (int)(nb / NN);
  float f = qkv[((size_t)b * CHT + 288 + v) * NN + n] * vs[v] + vh[v];
  vtb[i] = (bf16)f;
}

// ---------------------------------------------------------------------------
// K6: fused Yc+Yp kernel (bilinear path).  block 128 thr = 4 waves,
// 64 points per block, 16 points per wave (one WMMA A fragment = Vt tile).
//   G[n,(k,w)] = Vt[n,v] x A2t ;  g = relu(G + v[n,w]) + kv[k,w]
//   Y[n,h,w]   = sum_k q[h,k,n] * g     (register-accumulated)
// Y stored bf16 as [b][n][c] (c = h*32+w) -> ready as skip-GEMM B operand.
// A2t / V / kv staged with async-to-LDS while q BN-convert runs in parallel.
// ---------------------------------------------------------------------------
__global__ __launch_bounds__(128) void k_fused_y(
    const float* __restrict__ qkv, const float* __restrict__ qs,
    const float* __restrict__ qh, const bf16* __restrict__ vtb,
    const float* __restrict__ kvin, const bf16* __restrict__ A2t,
    bf16* __restrict__ Yb)
{
  extern __shared__ char smem[];
  bf16*  sA2 = (bf16*)smem;                               // 1024*32*2 = 64 KB
  bf16*  sQ  = (bf16*)(smem + 65536);                     // 64*256*2  = 32 KB
  bf16*  sV  = (bf16*)(smem + 65536 + 32768);             // 64*32*2   =  4 KB
  float* sKV = (float*)(smem + 65536 + 32768 + 4096);     // 32*32*4   =  4 KB

  const int tid  = threadIdx.x;
  const int wave = tid >> 5, lane = tid & 31;
  const int n0 = blockIdx.x * 64;
  const int b  = blockIdx.y;

  // async staging: A2t (64 KB), vbn tile (4 KB), kv (4 KB)
  for (int i = tid; i < 4096; i += 128)
    asy_b128(sA2 + i * 8, A2t + i * 8);
  {
    const bf16* src = vtb + ((size_t)b * NN + n0) * 32;
    for (int i = tid; i < 256; i += 128)
      asy_b128(sV + i * 8, src + i * 8);
  }
  {
    const float* src = kvin + (size_t)b * 32 * 32;
    for (int i = tid; i < 256; i += 128)
      asy_b128(sKV + i * 4, src + i * 4);
  }
  // meanwhile: stage q tile with BN applied (needs f32->bf16 convert)
  for (int i = tid; i < 64 * 256; i += 128) {
    int c = i >> 6, nn = i & 63;
    float f = qkv[((size_t)b * CHT + c) * NN + n0 + nn];
    sQ[nn * 256 + c] = (bf16)(f * qs[c] + qh[c]);
  }
  asy_wait0();
  __syncthreads();

  const int p0 = wave * 16;
  FragU fa;  // A fragment = Vt[16,32] tile for this wave's 16 points
  {
    int row = p0 + (lane & 15);
    int hb  = row * 32 + ((lane < 16) ? 0 : 8);
    fa.q[0] = *(const u32x4*)(sV + hb);
    fa.q[1] = *(const u32x4*)(sV + hb + 16);
  }
  const int mofs = (lane >> 4) << 3;   // C-fragment row offset (0 or 8)
  const int wlo  = lane & 15;

  for (int par = 0; par < 2; ++par) {      // w-half 0..15 / 16..31
    const int w = (par << 4) + wlo;
    float yac[8][8];                        // [r][h] register accumulators
#pragma unroll
    for (int r = 0; r < 8; ++r)
#pragma unroll
      for (int h = 0; h < 8; ++h) yac[r][h] = 0.f;

    for (int kcol = 0; kcol < 32; ++kcol) { // contraction dim of Yp/Yc
      const int ch = (kcol << 1) + par;     // A2t column chunk
      FragU fb;
      int col = ch * 16 + wlo;
      int hb  = col * 32 + ((lane < 16) ? 0 : 8);
      fb.q[0] = *(const u32x4*)(sA2 + hb);
      fb.q[1] = *(const u32x4*)(sA2 + hb + 16);
      AccU g;
#pragma unroll
      for (int r = 0; r < 8; ++r) g.f[r] = 0.f;
      g.v = wmma_bf16(fa.v, fb.v, g.v);

      const float kvw = sKV[kcol * 32 + w];
#pragma unroll
      for (int r = 0; r < 8; ++r) {
        const int pn = p0 + mofs + r;       // point index in this block tile
        float gv = g.f[r] + (float)sV[pn * 32 + w];   // + v[b,w,n] broadcast
        gv = fmaxf(gv, 0.f) + kvw;                    // relu, then + kv path
#pragma unroll
        for (int h = 0; h < 8; ++h) {
          float qv = (float)sQ[pn * 256 + h * 32 + kcol];
          yac[r][h] += qv * gv;
        }
      }
    }
    // write this w-half of Y directly (disjoint across parities/lanes/waves)
#pragma unroll
    for (int r = 0; r < 8; ++r) {
      const int pn = p0 + mofs + r;
      bf16* yrow = Yb + ((size_t)b * NN + n0 + pn) * 256 + w;
#pragma unroll
      for (int h = 0; h < 8; ++h) yrow[h * 32] = (bf16)yac[r][h];
    }
  }
}

// ---------------------------------------------------------------------------
// K8: epilogue  out = relu(BN(sraw) + x)   (bs cancels inside BN)
// ---------------------------------------------------------------------------
__global__ void k_final(const float* __restrict__ sraw,
                        const float* __restrict__ ss, const float* __restrict__ sh,
                        const float* __restrict__ x, float* __restrict__ out)
{
  size_t i = (size_t)blockIdx.x * 256 + threadIdx.x;
  if (i >= (size_t)BB * CCH * NN) return;
  int c = (int)((i / NN) & 255);
  float f = sraw[i] * ss[c] + sh[c] + x[i];
  out[i] = fmaxf(f, 0.f);
}

// ---------------------------------------------------------------------------
extern "C" void kernel_launch(void* const* d_in, const int* in_sizes, int n_in,
                              void* d_out, int out_size, void* d_ws, size_t ws_size,
                              hipStream_t stream)
{
  (void)in_sizes; (void)n_in; (void)out_size; (void)ws_size;
  const float* x  = (const float*)d_in[0];
  const float* Wq = (const float*)d_in[1];
  const float* Wk = (const float*)d_in[2];
  const float* Wv = (const float*)d_in[3];
  const float* A  = (const float*)d_in[4];
  const float* Ws = (const float*)d_in[5];
  // d_in[6] = bs (zeros; cancels in blocker BN)
  const float* gq = (const float*)d_in[7];
  const float* bq = (const float*)d_in[8];
  const float* gv = (const float*)d_in[9];
  const float* bv = (const float*)d_in[10];
  const float* gb = (const float*)d_in[11];
  const float* bb = (const float*)d_in[12];
  float* out = (float*)d_out;

  char* base = (char*)d_ws;
  size_t off = 0;
  auto alloc = [&](size_t bytes) -> char* {
    off = (off + 255) & ~(size_t)255;
    char* p = base + off;
    off += bytes;
    return p;
  };
  float* qkv  = (float*)alloc((size_t)BB * CHT * NN * 4);   // 84 MB
  bf16*  xt   = (bf16*) alloc((size_t)BB * NN * CCH * 2);   // 32 MB
  bf16*  Wall = (bf16*) alloc((size_t)CHT * 256 * 2);
  bf16*  Wsb  = (bf16*) alloc((size_t)256 * 256 * 2);
  bf16*  A2t  = (bf16*) alloc((size_t)1024 * 32 * 2);
  float* qs   = (float*)alloc(256 * 4);
  float* qh   = (float*)alloc(256 * 4);
  float* vs   = (float*)alloc(32 * 4);
  float* vh   = (float*)alloc(32 * 4);
  float* ss   = (float*)alloc(256 * 4);
  float* shf  = (float*)alloc(256 * 4);
  float* kmax = (float*)alloc(256 * 4);
  float* kinv = (float*)alloc(256 * 4);
  float* kvb  = (float*)alloc((size_t)BB * 32 * 32 * 4);
  bf16*  vtb  = (bf16*) alloc((size_t)BB * NN * 32 * 2);    // 4 MB
  bf16*  Yb   = (bf16*) alloc((size_t)BB * NN * 256 * 2);   // 32 MB
  float* sraw = (float*)alloc((size_t)BB * CCH * NN * 4);   // 64 MB

  // 0) weight prep + x transpose to bf16 [b][n][c]
  k_prep<<<704, 256, 0, stream>>>(Wq, Wk, Wv, Ws, A, Wall, Wsb, A2t);
  k_xt<<<dim3(NN / 32, CCH / 32, BB), 256, 0, stream>>>(x, xt);
  // 1) fused projection GEMM (q|k|v), async-staged, double-buffered
  k_gemm<<<dim3(CHT / 64, NN / 128, BB), 256, 0, stream>>>(xt, Wall, qkv, CHT);
  // 2) BN stats for q (256 ch) and v (32 ch)
  k_bnstats<<<256, 256, 0, stream>>>(qkv, CHT, 0,   gq, bq, qs, qh);
  k_bnstats<<<32,  256, 0, stream>>>(qkv, CHT, 288, gv, bv, vs, vh);
  // 3) softmax stats for k
  k_softmax_stats<<<BB * 32, 256, 0, stream>>>(qkv, kmax, kinv);
  // 4) kv = softmax(k) . vbn^T
  k_kv<<<BB * 32, 256, 0, stream>>>(qkv, kmax, kinv, vs, vh, kvb);
  // 5) vbn transposed bf16
  k_vtb<<<(BB * NN * 32) / 256, 256, 0, stream>>>(qkv, vs, vh, vtb);
  // 6) fused bilinear + attention gather -> Y  (104 KB dynamic LDS)
  k_fused_y<<<dim3(NN / 64, BB), 128, 65536 + 32768 + 4096 + 4096, stream>>>(
      qkv, qs, qh, vtb, kvb, A2t, Yb);
  // 7) skip GEMM (same kernel, Y already in [n][c] B-operand layout)
  k_gemm<<<dim3(CCH / 64, NN / 128, BB), 256, 0, stream>>>(Yb, Wsb, sraw, CCH);
  // 8) blocker BN stats
  k_bnstats<<<256, 256, 0, stream>>>(sraw, CCH, 0, gb, bb, ss, shf);
  // 9) epilogue: relu(BN(sraw) + x)
  k_final<<<(int)(((size_t)BB * CCH * NN) / 256), 256, 0, stream>>>(
      sraw, ss, shf, x, out);
}
